// RNNModel_20452634264070
// MI455X (gfx1250) — compile-verified, matrix operand order checked by reading, not run on hardware
//
#include <hip/hip_runtime.h>

typedef __attribute__((ext_vector_type(16))) _Float16 v16h;
typedef __attribute__((ext_vector_type(8)))  _Float16 v8h;
typedef __attribute__((ext_vector_type(8)))  float    v8f;

#define HID 128
#define SEQ 2048
#define BATCH 128
#define STR 136   // padded LDS row stride in halves (272B): kills ds bank conflicts

// Build a 16x32 f16 fragment (A or B operand) from an LDS row.
// ISA 7.12.2 layout: lanes 0-15 hold K = {kb..kb+7, kb+16..kb+23},
// lanes 16-31 hold K = {kb+8..kb+15, kb+24..kb+31}.
__device__ __forceinline__ v16h frag_from_lds(const _Float16* row, int kb, int hi) {
    const int K0 = kb + (hi ? 8 : 0);
    v8h s0 = *(const v8h*)(row + K0);        // ds_load_b128
    v8h s1 = *(const v8h*)(row + K0 + 16);   // ds_load_b128
    v16h r;
#pragma unroll
    for (int j = 0; j < 8; ++j) { r[j] = s0[j]; r[j + 8] = s1[j]; }
    return r;
}

// Same fragment built from a row-major f32 global row (one-time weight preload).
__device__ __forceinline__ v16h frag_from_global_f32(const float* __restrict__ row, int kb, int hi) {
    const int K0 = kb + (hi ? 8 : 0);
    v16h r;
#pragma unroll
    for (int j = 0; j < 8; ++j) {
        r[j]     = (_Float16)row[K0 + j];
        r[j + 8] = (_Float16)row[K0 + 16 + j];
    }
    return r;
}

__global__ __launch_bounds__(256, 1)
void rnn_persistent(const float* __restrict__ x,
                    const float* __restrict__ W_ih,
                    const float* __restrict__ W_hh,
                    const float* __restrict__ b_ih,
                    const float* __restrict__ b_hh,
                    float* __restrict__ out) {
    // Ping-pong state buffers: step with parity p reads [p], writes [1-p].
    __shared__ __align__(16) _Float16 xts[2][16 * STR];
    __shared__ __align__(16) _Float16 h0s[2][16 * STR];
    __shared__ __align__(16) _Float16 h1s[2][16 * STR];

    const int tid   = threadIdx.x;
    const int wid   = tid >> 5;        // 8 waves: wave w owns N in [16w, 16w+16)
    const int lane  = tid & 31;
    const int ln    = lane & 15;       // N (B/C frags) == M (A frags)
    const int hi    = lane >> 4;
    const int nBase = wid * 16;
    const int bBase = blockIdx.x * 16; // 8 blocks x 16 batch rows
    const int n     = nBase + ln;      // this lane's output hidden unit

    const float bias0 = b_ih[n]       + b_hh[n];
    const float bias1 = b_ih[HID + n] + b_hh[HID + n];

    // ---- one-time: weight B-fragments held in VGPRs (128 VGPRs/wave) ----
    v16h wih0[4], whh0[4], wih1[4], whh1[4];
    {
        const float* r_ih0 = W_ih + (size_t)n * HID;
        const float* r_hh0 = W_hh + (size_t)n * HID;
        const float* r_ih1 = W_ih + (size_t)HID * HID + (size_t)n * HID;
        const float* r_hh1 = W_hh + (size_t)HID * HID + (size_t)n * HID;
#pragma unroll
        for (int kc = 0; kc < 4; ++kc) {
            wih0[kc] = frag_from_global_f32(r_ih0, kc * 32, hi);
            whh0[kc] = frag_from_global_f32(r_hh0, kc * 32, hi);
            wih1[kc] = frag_from_global_f32(r_ih1, kc * 32, hi);
            whh1[kc] = frag_from_global_f32(r_hh1, kc * 32, hi);
        }
    }

    // Cooperative init: zero h (both buffers), load x_0 into buffer 0.
    const int rowc = tid >> 4;          // 0..15
    const int colc = (tid & 15) * 8;    // 0..120
    const float* xbase = x + (size_t)(bBase + rowc) * SEQ * HID + colc;
    {
#pragma unroll
        for (int j = 0; j < 8; ++j) {
            h0s[0][rowc * STR + colc + j] = (_Float16)0.f;
            h0s[1][rowc * STR + colc + j] = (_Float16)0.f;
            h1s[0][rowc * STR + colc + j] = (_Float16)0.f;
            h1s[1][rowc * STR + colc + j] = (_Float16)0.f;
        }
        v8h v;
#pragma unroll
        for (int j = 0; j < 8; ++j) v[j] = (_Float16)xbase[j];
        *(v8h*)(&xts[0][rowc * STR + colc]) = v;
    }
    __syncthreads();

    v8f c1acc = {};  // final-step layer-1 result (f32, survives the loop)

    // Dual-operand GEMM tile: relu(A0 @ W0^T + A1 @ W1^T + bias), K=128.
    // A-fragment loads pipelined one K-chunk ahead of the WMMA chain.
    auto gemm2 = [&](const _Float16* rA0, const _Float16* rA1,
                     const v16h* w0, const v16h* w1, float bias) -> v8f {
        v8f cx = {}, ch;
#pragma unroll
        for (int i = 0; i < 8; ++i) ch[i] = bias;
        v16h a0 = frag_from_lds(rA0, 0, hi);
        v16h a1 = frag_from_lds(rA1, 0, hi);
#pragma unroll
        for (int kc = 0; kc < 4; ++kc) {
            v16h p0, p1;
            if (kc < 3) {
                p0 = frag_from_lds(rA0, (kc + 1) * 32, hi);
                p1 = frag_from_lds(rA1, (kc + 1) * 32, hi);
            }
            cx = __builtin_amdgcn_wmma_f32_16x16x32_f16(false, a0, false, w0[kc],
                                                        (short)0, cx, false, false);
            ch = __builtin_amdgcn_wmma_f32_16x16x32_f16(false, a1, false, w1[kc],
                                                        (short)0, ch, false, false);
            if (kc < 3) { a0 = p0; a1 = p1; }
        }
        v8f r;
#pragma unroll
        for (int i = 0; i < 8; ++i) r[i] = fmaxf(cx[i] + ch[i], 0.f);
        return r;
    };

    // One RNN step, single barrier (ping-pong removes the other three).
    auto step = [&](int t, int p) {
        const int q = 1 - p;
        const _Float16* xinR  = &xts[p][ln * STR];
        const _Float16* h0inR = &h0s[p][ln * STR];
        const _Float16* h1inR = &h1s[p][ln * STR];
        const _Float16* h0nR  = &h0s[q][ln * STR];

        // Register-prefetch x_{t+1}; completes under the layer-0 WMMAs.
        v8h pf;
        const bool more = (t + 1 < SEQ);
        if (more) {
            const float* xp = xbase + (size_t)(t + 1) * HID;
#pragma unroll
            for (int j = 0; j < 8; ++j) pf[j] = (_Float16)xp[j];
        }

        // layer 0: h0' = relu(x_t @ Wih0^T + h0 @ Whh0^T + b0)
        v8f h0n = gemm2(xinR, h0inR, wih0, whh0, bias0);
        // C layout: lane ln = N column, vgpr i -> batch row M = i + 8*hi
#pragma unroll
        for (int i = 0; i < 8; ++i) h0s[q][(i + hi * 8) * STR + n] = (_Float16)h0n[i];
        if (more) *(v8h*)(&xts[q][rowc * STR + colc]) = pf;  // publish x_{t+1}

        __syncthreads();  // sole barrier: h0'/x_{t+1} visible, h1[p] reads fenced

        // layer 1: h1' = relu(h0' @ Wih1^T + h1 @ Whh1^T + b1)
        v8f h1n = gemm2(h0nR, h1inR, wih1, whh1, bias1);
#pragma unroll
        for (int i = 0; i < 8; ++i) h1s[q][(i + hi * 8) * STR + n] = (_Float16)h1n[i];
        c1acc = h1n;
    };

    for (int t = 0; t < SEQ; t += 2) {  // SEQ even: parity is compile-time
        step(t,     0);
        step(t + 1, 1);
    }

    // Output h1[:, 3] in f32 straight from accumulator registers:
    // N=3 lives in wave 0, lanes 3 (rows 0-7) and 19 (rows 8-15).
    if (wid == 0 && ln == 3) {
#pragma unroll
        for (int i = 0; i < 8; ++i) out[bBase + i + hi * 8] = c1acc[i];
    }
}

extern "C" void kernel_launch(void* const* d_in, const int* in_sizes, int n_in,
                              void* d_out, int out_size, void* d_ws, size_t ws_size,
                              hipStream_t stream) {
    (void)in_sizes; (void)n_in; (void)d_ws; (void)ws_size; (void)out_size;
    const float* x    = (const float*)d_in[0];
    const float* W_ih = (const float*)d_in[1];
    const float* W_hh = (const float*)d_in[2];
    const float* b_ih = (const float*)d_in[3];
    const float* b_hh = (const float*)d_in[4];
    float* out = (float*)d_out;
    hipLaunchKernelGGL(rnn_persistent, dim3(BATCH / 16), dim3(256), 0, stream,
                       x, W_ih, W_hh, b_ih, b_hh, out);
}
